// SUREL_63367947485756
// MI455X (gfx1250) — matrix operand order repaired and checked by reading, not compiled, not run on hardware
//
#include <hip/hip_runtime.h>
#include <hip/hip_bf16.h>

typedef __attribute__((ext_vector_type(16))) _Float16 v16h;
typedef __attribute__((ext_vector_type(8)))  float    v8f;

#define NN      10000
#define NSAMP   2048
#define NWALK   50
#define WLEN    4
#define EDIM    64
#define B2DIM   (2 * NSAMP)           // 4096
#define MROWS   (B2DIM * NWALK)       // 204800 sequence rows
#define NTILES  (MROWS / 16)          // 12800
#define WAVES_PER_BLOCK 4

// ---- workspace layout (halves / bytes) ----
// Wpack: 77824 halves  (W1p 4096 | W2 8192 | Wih0 16384 | Whh0 16384 | Wih1 16384 | Whh1 16384)
#define WPACK_HALVES 77824
#define OFF_W1   0
#define OFF_W2   4096
#define OFF_WIH0 12288
#define OFF_WHH0 28672
#define OFF_WIH1 45056
#define OFF_WHH1 61440
#define X_OFF_BYTES   (WPACK_HALVES * 2)                       // f16 [4][204800][8]
#define X_HALVES      (4 * MROWS * 8)
#define Z_OFF_BYTES   (X_OFF_BYTES + X_HALVES * 2)             // f32 [204800][64]

union V8  { v8f  v; float f[8]; };
union F16x16 { v16h v; uint4 q[2]; };

// Fast device transcendentals: raw v_rcp_f32 + v_exp_f32 (no IEEE div fixup,
// no ocml tanh call). Error far below the f16-WMMA quantization already used.
__device__ __forceinline__ float sigf(float x) {
    return __builtin_amdgcn_rcpf(1.0f + __expf(-x));
}
__device__ __forceinline__ float tanh_fast(float x) {
    return fmaf(2.0f, __builtin_amdgcn_rcpf(1.0f + __expf(-2.0f * x)), -1.0f);
}

// Load 16x32 f16 A-fragment from row-major LDS activation [16][K].
// ISA layout: lanes 0-15 (row m=lane): K = kb*32 + {0..7, 16..23}; lanes 16-31: +8.
__device__ __forceinline__ v16h ldsA(const _Float16* act, int K, int kb, int lane) {
    int m = lane & 15, g = lane >> 4;
    const _Float16* p = act + m * K + kb * 32 + g * 8;
    F16x16 u;
    u.q[0] = *(const uint4*)p;
    u.q[1] = *(const uint4*)(p + 16);
    return u.v;
}

// Load prepacked 32x16 B-fragment from global: frag fid, lane-major 16 halves each.
__device__ __forceinline__ v16h gblB(const _Float16* w, int fid, int lane) {
    const _Float16* p = w + (size_t)fid * 512 + lane * 16;
    F16x16 u;
    u.q[0] = *(const uint4*)p;
    u.q[1] = *(const uint4*)(p + 8);
    return u.v;
}

__device__ __forceinline__ v8f wmma16(v16h a, v16h b, v8f c) {
    return __builtin_amdgcn_wmma_f32_16x16x32_f16(false, a, false, b, (short)0, c, false, false);
}

// ---------------- kernel 0: weight prepack into B-fragment layout ----------------
__global__ void SUREL_prepack(const float* W1, const float* W2,
                              const float* Wih0, const float* Whh0,
                              const float* Wih1, const float* Whh1, _Float16* dstw) {
    int tid = blockIdx.x * 256 + threadIdx.x;
    if (tid >= WPACK_HALVES) return;
    const float* src; int Kreal, Kpad, Nmat, off;
    if      (tid < OFF_W2)   { src = W1;   Kreal = 8;   Kpad = 32;  Nmat = 128; off = OFF_W1; }
    else if (tid < OFF_WIH0) { src = W2;   Kreal = 128; Kpad = 128; Nmat = 64;  off = OFF_W2; }
    else if (tid < OFF_WHH0) { src = Wih0; Kreal = 64;  Kpad = 64;  Nmat = 256; off = OFF_WIH0; }
    else if (tid < OFF_WIH1) { src = Whh0; Kreal = 64;  Kpad = 64;  Nmat = 256; off = OFF_WHH0; }
    else if (tid < OFF_WHH1) { src = Wih1; Kreal = 64;  Kpad = 64;  Nmat = 256; off = OFF_WIH1; }
    else                     { src = Whh1; Kreal = 64;  Kpad = 64;  Nmat = 256; off = OFF_WHH1; }
    int h = tid - off;
    int fid = h >> 9, rem = h & 511, lane = rem >> 4, i = rem & 15;
    int KB = Kpad >> 5;
    int kb = fid % KB, nb = fid / KB;
    int n = nb * 16 + (lane & 15);
    int k = kb * 32 + (lane >> 4) * 8 + (i & 7) + (i >> 3) * 16;
    float v = (k < Kreal) ? src[k * Nmat + n] : 0.0f;
    dstw[tid] = (_Float16)v;
}

// ---------------- kernel 1: gather/embedding -> X f16 [4][204800][8] ----------------
__global__ void SUREL_embed(const int* walks, const int* Hm, const float* T,
                            const int* src, const int* dst, _Float16* X) {
    int tid = blockIdx.x * 256 + threadIdx.x;
    if (tid >= MROWS * 4) return;
    int t = tid & 3, row = tid >> 2;
    int b2 = row / NWALK, j = row - b2 * NWALK;
    int bi = b2 & (NSAMP - 1);
    int batch = (b2 < NSAMP) ? src[bi] : dst[bi];
    int au = src[bi], av = dst[bi];
    int wn = walks[(batch * NWALK + j) * WLEN + t];
    const float* Tu = T + Hm[au * NN + wn] * WLEN;
    const float* Tv = T + Hm[av * NN + wn] * WLEN;
    _Float16* xp = X + ((size_t)t * MROWS + row) * 8;
#pragma unroll
    for (int k = 0; k < 4; ++k) { xp[k] = (_Float16)Tu[k]; xp[4 + k] = (_Float16)Tv[k]; }
}

// ---------------- kernel 2: fused MLP + 2-layer LSTM over 4 timesteps ----------------
__device__ __forceinline__ v8f gate4(v16h ax0, v16h ax1, v16h ah0, v16h ah1,
                                     const _Float16* wih, const _Float16* whh,
                                     float bv, int nt, int lane) {
    v8f acc = {bv, bv, bv, bv, bv, bv, bv, bv};
    acc = wmma16(ax0, gblB(wih, nt * 2 + 0, lane), acc);
    acc = wmma16(ax1, gblB(wih, nt * 2 + 1, lane), acc);
    acc = wmma16(ah0, gblB(whh, nt * 2 + 0, lane), acc);
    acc = wmma16(ah1, gblB(whh, nt * 2 + 1, lane), acc);
    return acc;
}

__device__ void lstm_step(const _Float16* xact, const _Float16* hcur, _Float16* hnext,
                          const float* bihp, const float* bhhp,
                          const _Float16* wih, const _Float16* whh, V8* c, int lane) {
    int g = lane >> 4, nl = lane & 15;
    v16h ax0 = ldsA(xact, 64, 0, lane), ax1 = ldsA(xact, 64, 1, lane);
    v16h ah0 = ldsA(hcur, 64, 0, lane), ah1 = ldsA(hcur, 64, 1, lane);
#pragma unroll
    for (int j = 0; j < 4; ++j) {
        V8 gi, gf, gg, go;
        gi.v = gate4(ax0, ax1, ah0, ah1, wih, whh, bihp[(j     ) * 16 + nl] + bhhp[(j     ) * 16 + nl], j,      lane);
        gf.v = gate4(ax0, ax1, ah0, ah1, wih, whh, bihp[(j +  4) * 16 + nl] + bhhp[(j +  4) * 16 + nl], j +  4, lane);
        gg.v = gate4(ax0, ax1, ah0, ah1, wih, whh, bihp[(j +  8) * 16 + nl] + bhhp[(j +  8) * 16 + nl], j +  8, lane);
        go.v = gate4(ax0, ax1, ah0, ah1, wih, whh, bihp[(j + 12) * 16 + nl] + bhhp[(j + 12) * 16 + nl], j + 12, lane);
#pragma unroll
        for (int r = 0; r < 8; ++r) {
            float iv = sigf(gi.f[r]);
            float fv = sigf(gf.f[r]);
            float gv = tanh_fast(gg.f[r]);
            float ov = sigf(go.f[r]);
            float cv = fmaf(fv, c[j].f[r], iv * gv);
            c[j].f[r] = cv;
            hnext[(r + 8 * g) * 64 + j * 16 + nl] = (_Float16)(ov * tanh_fast(cv));
        }
    }
}

// Cap at 256 VGPRs (min 4 waves/EU): removes s_set_vgpr_msb/v_nop overhead and
// keeps 4 waves/SIMD for latency hiding; small scratch spill is acceptable.
__global__ void __launch_bounds__(WAVES_PER_BLOCK * 32, 4)
SUREL_pipeline(const _Float16* X, const _Float16* Wpack,
               const float* b1, const float* b2v,
               const float* bih0, const float* bhh0,
               const float* bih1, const float* bhh1, float* Z) {
    extern __shared__ _Float16 smem[];
    int wave = threadIdx.x >> 5, lane = threadIdx.x & 31;
    int tile = blockIdx.x * WAVES_PER_BLOCK + wave;
    int g = lane >> 4, nl = lane & 15;

    _Float16* act1 = smem + wave * 7168;     // 16x128
    _Float16* act2 = act1 + 2048;            // 16x64
    _Float16* h0a  = act2 + 1024;
    _Float16* h0b  = h0a  + 1024;
    _Float16* h1a  = h0b  + 1024;
    _Float16* h1b  = h1a  + 1024;

    for (int i = lane; i < 1024; i += 32) { h0a[i] = (_Float16)0.0f; h1a[i] = (_Float16)0.0f; }
    V8 c0[4] = {}, c1[4] = {};

    const _Float16* W1f   = Wpack + OFF_W1;
    const _Float16* W2f   = Wpack + OFF_W2;
    const _Float16* Wih0f = Wpack + OFF_WIH0;
    const _Float16* Whh0f = Wpack + OFF_WHH0;
    const _Float16* Wih1f = Wpack + OFF_WIH1;
    const _Float16* Whh1f = Wpack + OFF_WHH1;
    __builtin_prefetch(Wpack + (size_t)lane * 512, 0, 3);   // warm WGP$/L2 with weights

    _Float16 *h0c = h0a, *h0n = h0b, *h1c = h1a, *h1n = h1b;

    for (int t = 0; t < 4; ++t) {
        // ---- A-fragment of x_t (16x8 zero-padded to 16x32), straight from global X
        F16x16 ua;
        ua.q[0] = make_uint4(0u, 0u, 0u, 0u);
        ua.q[1] = make_uint4(0u, 0u, 0u, 0u);
        if (lane < 16)
            ua.q[0] = *(const uint4*)(X + (((size_t)t * MROWS) + (size_t)tile * 16 + lane) * 8);
        v16h ax = ua.v;

        // ---- MLP1: 16x32 @ 32x128 (K padded), relu -> act1
#pragma unroll
        for (int nb = 0; nb < 8; ++nb) {
            float bv = b1[nb * 16 + nl];
            v8f acc = {bv, bv, bv, bv, bv, bv, bv, bv};
            acc = wmma16(ax, gblB(W1f, nb, lane), acc);
            V8 o; o.v = acc;
#pragma unroll
            for (int r = 0; r < 8; ++r)
                act1[(r + 8 * g) * 128 + nb * 16 + nl] = (_Float16)fmaxf(o.f[r], 0.0f);
        }

        // ---- MLP2: 16x128 @ 128x64, relu -> act2
        {
            v16h a[4];
#pragma unroll
            for (int kb = 0; kb < 4; ++kb) a[kb] = ldsA(act1, 128, kb, lane);
#pragma unroll
            for (int nb = 0; nb < 4; ++nb) {
                float bv = b2v[nb * 16 + nl];
                v8f acc = {bv, bv, bv, bv, bv, bv, bv, bv};
#pragma unroll
                for (int kb = 0; kb < 4; ++kb)
                    acc = wmma16(a[kb], gblB(W2f, nb * 4 + kb, lane), acc);
                V8 o; o.v = acc;
#pragma unroll
                for (int r = 0; r < 8; ++r)
                    act2[(r + 8 * g) * 64 + nb * 16 + nl] = (_Float16)fmaxf(o.f[r], 0.0f);
            }
        }

        // ---- LSTM layers
        lstm_step(act2, h0c, h0n, bih0, bhh0, Wih0f, Whh0f, c0, lane);
        lstm_step(h0n,  h1c, h1n, bih1, bhh1, Wih1f, Whh1f, c1, lane);
        _Float16* tmp;
        tmp = h0c; h0c = h0n; h0n = tmp;
        tmp = h1c; h1c = h1n; h1n = tmp;
    }

    // ---- emit z = final hidden of LSTM1 (16x64) to global as f32
    for (int i = lane; i < 1024; i += 32) {
        int m = i >> 6, e = i & 63;
        Z[((size_t)tile * 16 + m) * 64 + e] = (float)h1c[i];
    }
}

// ---------------- kernel 3: walk-mean + src/dst combine + head MLP ----------------
__global__ void SUREL_head(const float* Z, const float* lpW1, const float* lpb1,
                           const float* lpW2, const float* lpb2, float* out) {
    __shared__ float zc[64];
    __shared__ float part[64];
    int i = blockIdx.x, e = threadIdx.x;      // 2048 blocks x 64 threads
    float s = 0.0f;
    for (int j = 0; j < NWALK; ++j) s += Z[((size_t)(i * NWALK + j)) * 64 + e];
    for (int j = 0; j < NWALK; ++j) s += Z[((size_t)((i + NSAMP) * NWALK + j)) * 64 + e];
    zc[e] = s * (0.5f / (float)NWALK);
    __syncthreads();
    float a = lpb1[e];
    for (int k = 0; k < 64; ++k) a += zc[k] * lpW1[k * 64 + e];
    part[e] = fmaxf(a, 0.0f) * lpW2[e];
    __syncthreads();
    for (int st = 32; st > 0; st >>= 1) {
        if (e < st) part[e] += part[e + st];
        __syncthreads();
    }
    if (e == 0) out[i] = part[0] + lpb2[0];
}

extern "C" void kernel_launch(void* const* d_in, const int* in_sizes, int n_in,
                              void* d_out, int out_size, void* d_ws, size_t ws_size,
                              hipStream_t stream) {
    const int*   walks = (const int*)d_in[0];
    const int*   Hm    = (const int*)d_in[1];
    const float* T     = (const float*)d_in[2];
    const int*   src   = (const int*)d_in[3];
    const int*   dst   = (const int*)d_in[4];
    const float* W1    = (const float*)d_in[5];
    const float* b1    = (const float*)d_in[6];
    const float* W2    = (const float*)d_in[7];
    const float* b2v   = (const float*)d_in[8];
    const float* Wih0  = (const float*)d_in[9];
    const float* Whh0  = (const float*)d_in[10];
    const float* bih0  = (const float*)d_in[11];
    const float* bhh0  = (const float*)d_in[12];
    const float* Wih1  = (const float*)d_in[13];
    const float* Whh1  = (const float*)d_in[14];
    const float* bih1  = (const float*)d_in[15];
    const float* bhh1  = (const float*)d_in[16];
    const float* lpW1  = (const float*)d_in[17];
    const float* lpb1  = (const float*)d_in[18];
    const float* lpW2  = (const float*)d_in[19];
    const float* lpb2  = (const float*)d_in[20];

    char* ws = (char*)d_ws;
    _Float16* Wpack = (_Float16*)(ws);
    _Float16* X     = (_Float16*)(ws + X_OFF_BYTES);
    float*    Z     = (float*)   (ws + Z_OFF_BYTES);

    SUREL_prepack<<<(WPACK_HALVES + 255) / 256, 256, 0, stream>>>(W1, W2, Wih0, Whh0, Wih1, Whh1, Wpack);
    SUREL_embed<<<(MROWS * 4 + 255) / 256, 256, 0, stream>>>(walks, Hm, T, src, dst, X);
    SUREL_pipeline<<<NTILES / WAVES_PER_BLOCK, WAVES_PER_BLOCK * 32,
                     WAVES_PER_BLOCK * 7168 * sizeof(_Float16), stream>>>(
        X, Wpack, b1, b2v, bih0, bhh0, bih1, bhh1, Z);
    SUREL_head<<<NSAMP, 64, 0, stream>>>(Z, lpW1, lpb1, lpW2, lpb2, (float*)d_out);
}
